// CausalMultiHeadAttention_33827162423716
// MI455X (gfx1250) — matrix operand order repaired and checked by reading, not compiled, not run on hardware
//
#include <hip/hip_runtime.h>

// ---------------------------------------------------------------------------
// Causal multi-head attention, MI455X (gfx1250, wave32, WMMA bf16->f32)
//   x:[2,2048,1024] f32, Wq/Wk/Wv/Wo:[1024,1024] f32  ->  out:[2,2048,1024] f32
// Pipeline: bf16 convert -> fused QKV GEMM (TDM double-buffered) ->
//           flash attention -> out GEMM
// ---------------------------------------------------------------------------

#define S_LEN   2048
#define D_MODEL 1024
#define NHEADS  16
#define DK      64
#define BATCH   2
#define NTOK    (BATCH * S_LEN)      // 4096

typedef __attribute__((ext_vector_type(16))) __bf16 v16bf;
typedef __attribute__((ext_vector_type(8)))  float  v8f;
typedef unsigned int __attribute__((ext_vector_type(4))) tdm_u32x4;
typedef int          __attribute__((ext_vector_type(4))) tdm_i32x4;
typedef int          __attribute__((ext_vector_type(8))) tdm_i32x8;

union FragB16 { v16bf v; unsigned u[8]; unsigned short s[16]; };
union FragF32 { v8f   v; float    f[8]; };

// TDM availability / arity discrimination (per CDNA5_HIP.md):
//  - builtin exists on both toolchains; therock headers ship the 6-arg form.
#if defined(__has_builtin)
#if __has_builtin(__builtin_amdgcn_tensor_load_to_lds)
#define USE_TDM 1
#else
#define USE_TDM 0
#endif
#else
#define USE_TDM 0
#endif
#if __has_include(<hip/amd_detail/amd_gfx1250_TDM.h>)
#define TDM_6ARG 1
#else
#define TDM_6ARG 0
#endif

__device__ __forceinline__ unsigned short f32_to_bf16(float f) {
  unsigned u = __float_as_uint(f);
  u += 0x7FFFu + ((u >> 16) & 1u);           // round-to-nearest-even
  return (unsigned short)(u >> 16);
}

__device__ __forceinline__ unsigned lds_off(const void* p) {
  // generic -> LDS(AS3) addrspace cast, then 32-bit LDS byte offset
  return (unsigned)(unsigned long long)(__attribute__((address_space(3))) const void*)p;
}

#if USE_TDM
// Issue one TDM 2-D tile load: global (elems, 2B each, row stride in elems)
// -> LDS with hardware row padding 64B data + 16B pad = 80B rows (LSTR=40).
__device__ __forceinline__ void tdm_tile_load(unsigned long long gaddr_bytes,
                                              unsigned lds_addr,
                                              unsigned tile_w, unsigned tile_h,
                                              unsigned tensor_w, unsigned tensor_h,
                                              unsigned row_stride_elems) {
  tdm_u32x4 g0;
  g0.x = 1u;                                             // count=1 (valid), user
  g0.y = lds_addr;                                       // LDS byte address
  g0.z = (unsigned)(gaddr_bytes & 0xFFFFFFFFull);        // global_addr[31:0]
  g0.w = (unsigned)((gaddr_bytes >> 32) & 0x1FFFFFFull)  // global_addr[56:32]
         | (2u << 30);                                   // type=2 ("image")
  tdm_i32x8 g1;
  g1[0] = (1 << 16)        // data_size = 2 bytes
        | (1 << 20)        // pad_enable
        | (3 << 22)        // pad_interval: 16 DWORDs (=64B, one 32-elem row)
        | (3 << 25);       // pad_amount:   4 DWORDs (=16B) -> 80B LDS rows
  g1[1] = (int)((tensor_w & 0xFFFFu) << 16);             // tensor_dim0[15:0]
  g1[2] = (int)((tensor_w >> 16) | ((tensor_h & 0xFFFFu) << 16));
  g1[3] = (int)((tensor_h >> 16) | (tile_w << 16));      // tile_dim0
  g1[4] = (int)tile_h;                                   // tile_dim1 (tile_dim2=0)
  g1[5] = (int)row_stride_elems;                         // tensor_dim0_stride lo
  g1[6] = 0;
  g1[7] = 0;
  tdm_i32x4 z4 = {0, 0, 0, 0};
#if TDM_6ARG
  tdm_i32x8 z8 = {0, 0, 0, 0, 0, 0, 0, 0};
  __builtin_amdgcn_tensor_load_to_lds(g0, g1, z4, z4, z8, 0);
#else
  __builtin_amdgcn_tensor_load_to_lds(g0, g1, z4, z4, 0);
#endif
}
#endif  // USE_TDM

// ---------------------------------------------------------------------------
// Stage 0: f32 -> bf16 conversion
// ---------------------------------------------------------------------------
__global__ void cvt_bf16_kernel(const float* __restrict__ src,
                                unsigned short* __restrict__ dst, int n) {
  int i = blockIdx.x * blockDim.x + threadIdx.x;
  if (i < n) dst[i] = f32_to_bf16(src[i]);
}

// ---------------------------------------------------------------------------
// Stages 1 & 3: C[M,N] = A[M,K](bf16) * W[N,K](bf16)^T
// Block: 256 thr = 8 waves; block tile 128(M) x 128(N); wave tile 32x64.
// Double-buffered LDS tiles; staging via TDM (wave 0 issues, TENSORcnt waits)
// with cooperative-copy fallback. LDS rows padded to 40 shorts (80B).
// ---------------------------------------------------------------------------
template <bool OUT_F32>
__global__ __launch_bounds__(256)
void gemm_bf16_wT_kernel(const unsigned short* __restrict__ A,
                         const unsigned short* __restrict__ W,
                         void* __restrict__ Cout, int M, int N, int K) {
  constexpr int LSTR = 40;
  __shared__ __align__(16) unsigned short As[2][128 * LSTR];
  __shared__ __align__(16) unsigned short Bs[2][128 * LSTR];

  const int tid  = threadIdx.x;
  const int lane = tid & 31;
  const int wave = tid >> 5;
  const int wm   = wave & 3;           // 4 M-subtiles of 32
  const int wn   = wave >> 2;          // 2 N-subtiles of 64
  const int lo   = lane & 15;
  const int hi   = lane >> 4;
  const int mbase = blockIdx.y * 128;
  const int nbase = blockIdx.x * 128;

  v8f acc[2][4];
#pragma unroll
  for (int i = 0; i < 2; ++i)
#pragma unroll
    for (int j = 0; j < 4; ++j) acc[i][j] = (v8f)0.0f;

  // ---- tile staging (TDM or cooperative fallback) ----
  auto stage = [&](int buf, int k0) {
#if USE_TDM
    if (wave == 0) {
      tdm_tile_load((unsigned long long)(const void*)A +
                        2ull * ((unsigned long long)mbase * K + k0),
                    lds_off(&As[buf][0]), 32, 128, (unsigned)K, (unsigned)M,
                    (unsigned)K);
      tdm_tile_load((unsigned long long)(const void*)W +
                        2ull * ((unsigned long long)nbase * K + k0),
                    lds_off(&Bs[buf][0]), 32, 128, (unsigned)K, (unsigned)N,
                    (unsigned)K);
    }
#else
    const int row = tid >> 1, half = tid & 1;
    {
      const unsigned short* g = A + (size_t)(mbase + row) * K + k0 + half * 16;
      __builtin_prefetch(g + 32, 0, 0);
      const uint4* g4 = (const uint4*)g;
      uint4* l4 = (uint4*)&As[buf][row * LSTR + half * 16];
      l4[0] = g4[0];
      l4[1] = g4[1];
    }
    {
      const unsigned short* g = W + (size_t)(nbase + row) * K + k0 + half * 16;
      __builtin_prefetch(g + 32, 0, 0);
      const uint4* g4 = (const uint4*)g;
      uint4* l4 = (uint4*)&Bs[buf][row * LSTR + half * 16];
      l4[0] = g4[0];
      l4[1] = g4[1];
    }
#endif
  };

  const int nsteps = K / 32;
  stage(0, 0);                                   // prologue: fill buffer 0

  for (int step = 0; step < nsteps; ++step) {
    const int cur = step & 1;
    __syncthreads();   // prev compute done -> safe to overwrite buf cur^1
    if (step + 1 < nsteps) stage(cur ^ 1, (step + 1) * 32);
#if USE_TDM
    if (wave == 0) {   // wait for current tile's DMA (2 in flight per stage)
      if (step + 1 < nsteps) __builtin_amdgcn_s_wait_tensorcnt(2);
      else                   __builtin_amdgcn_s_wait_tensorcnt(0);
    }
#endif
    __syncthreads();   // current buffer fully resident for all waves

    // A fragments (lane holds row lo, contiguous 16 K at hi*16)
    FragB16 a[2], b[4];
#pragma unroll
    for (int i = 0; i < 2; ++i) {
      const int row = wm * 32 + i * 16 + lo;
#pragma unroll
      for (int j = 0; j < 8; ++j) {
        const int kk = hi * 16 + j * 2;
        a[i].u[j] = *(const unsigned*)&As[cur][row * LSTR + kk];
      }
    }
    // B fragments (lane holds col lo, contiguous 16 K at hi*16)
#pragma unroll
    for (int j4 = 0; j4 < 4; ++j4) {
      const int row = wn * 64 + j4 * 16 + lo;
      const unsigned short* base = &Bs[cur][row * LSTR + hi * 16];
#pragma unroll
      for (int j = 0; j < 8; ++j) b[j4].u[j] = *(const unsigned*)(base + 2 * j);
    }
#pragma unroll
    for (int i = 0; i < 2; ++i)
#pragma unroll
      for (int j = 0; j < 4; ++j)
        acc[i][j] = __builtin_amdgcn_wmma_f32_16x16x32_bf16(
            false, a[i].v, false, b[j].v, (short)0, acc[i][j], false, false);
  }

  // C layout: dword r -> row r + 8*hi, col lo
#pragma unroll
  for (int i = 0; i < 2; ++i)
#pragma unroll
    for (int j = 0; j < 4; ++j) {
      FragF32 c;
      c.v = acc[i][j];
#pragma unroll
      for (int r = 0; r < 8; ++r) {
        const int row = mbase + wm * 32 + i * 16 + r + 8 * hi;
        const int col = nbase + wn * 64 + j * 16 + lo;
        if (OUT_F32)
          ((float*)Cout)[(size_t)row * N + col] = c.f[r];
        else
          ((unsigned short*)Cout)[(size_t)row * N + col] = f32_to_bf16(c.f[r]);
      }
    }
}

// ---------------------------------------------------------------------------
// Stage 2: causal flash attention.
// Grid: (S/64, B*H). Block: 128 thr = 4 waves, wave owns 16 query rows.
// Key tiles of 32; K tile [32][64] + V^T tile [64][32] staged in LDS.
// ---------------------------------------------------------------------------
__global__ __launch_bounds__(128)
void flash_attn_kernel(const unsigned short* __restrict__ qkv,   // [NTOK][3*D]
                       unsigned short* __restrict__ Obuf) {      // [NTOK][D]
  constexpr int LSTR = 40;
  __shared__ __align__(16) unsigned short Ks[32 * LSTR];   // [key][d]
  __shared__ __align__(16) unsigned short Vt[64 * LSTR];   // [d][key]
  __shared__ __align__(16) unsigned short Ps[4][16 * LSTR];

  const int tid  = threadIdx.x;
  const int lane = tid & 31;
  const int wave = tid >> 5;
  const int lo   = lane & 15;
  const int hi   = lane >> 4;
  const int bh   = blockIdx.y;
  const int b    = bh / NHEADS;
  const int h    = bh % NHEADS;
  const int q0b  = blockIdx.x * 64;       // block's first query row
  const int q0w  = q0b + wave * 16;       // wave's first query row
  const size_t rstride = 3 * D_MODEL;

  const unsigned short* Qg = qkv + (size_t)(b * S_LEN) * rstride + h * DK;
  const unsigned short* Kg = Qg + D_MODEL;
  const unsigned short* Vg = Qg + 2 * D_MODEL;

  // Q fragments: 2 dk-chunks of 32, loaded straight from global in A layout
  FragB16 qf[2];
  {
    const unsigned short* qr = Qg + (size_t)(q0w + lo) * rstride;
#pragma unroll
    for (int c = 0; c < 2; ++c)
#pragma unroll
      for (int j = 0; j < 8; ++j) {
        const int kk = c * 32 + hi * 16 + j * 2;
        qf[c].u[j] = *(const unsigned*)(qr + kk);
      }
  }

  v8f o[4];
#pragma unroll
  for (int c = 0; c < 4; ++c) o[c] = (v8f)0.0f;
  float mrow[8], lrow[8];
#pragma unroll
  for (int r = 0; r < 8; ++r) { mrow[r] = -__builtin_inff(); lrow[r] = 0.0f; }

  const int ntiles = (q0b + 64) / 32;
  for (int t = 0; t < ntiles; ++t) {
    const int j0 = t * 32;
    __syncthreads();
    {  // cooperative staging: thread -> key row tid>>2, 16-col segment
      const int kr  = tid >> 2;
      const int seg = (tid & 3) * 16;
      const unsigned short* gk = Kg + (size_t)(j0 + kr) * rstride + seg;
      const uint4* g4 = (const uint4*)gk;
      uint4 d0 = g4[0], d1 = g4[1];
      uint4* l4 = (uint4*)&Ks[kr * LSTR + seg];
      l4[0] = d0;
      l4[1] = d1;
      const unsigned short* gv = Vg + (size_t)(j0 + kr) * rstride + seg;
#pragma unroll
      for (int e = 0; e < 16; ++e) Vt[(seg + e) * LSTR + kr] = gv[e];
    }
    __syncthreads();

    if (j0 > q0w + 15) continue;   // wave-uniform causal skip (no barriers below)

    // S = Q K^T : two 16x16 score frags (key cols 0-15, 16-31)
    v8f s[2];
#pragma unroll
    for (int c = 0; c < 2; ++c) {
      s[c] = (v8f)0.0f;
#pragma unroll
      for (int ch = 0; ch < 2; ++ch) {
        FragB16 kb;
        const unsigned short* base = &Ks[(c * 16 + lo) * LSTR + ch * 32 + hi * 16];
#pragma unroll
        for (int j = 0; j < 8; ++j) kb.u[j] = *(const unsigned*)(base + 2 * j);
        s[c] = __builtin_amdgcn_wmma_f32_16x16x32_bf16(
            false, qf[ch].v, false, kb.v, (short)0, s[c], false, false);
      }
    }

    // online softmax (row stats reduced over 16-lane halves, matching C layout)
    FragF32 f0, f1;
    f0.v = s[0];
    f1.v = s[1];
    const float scale = 0.125f;   // 1/sqrt(64)
#pragma unroll
    for (int r = 0; r < 8; ++r) {
      const int qrow = q0w + r + 8 * hi;
      float a0 = f0.f[r] * scale;
      float a1 = f1.f[r] * scale;
      if (j0 + lo > qrow)      a0 = -__builtin_inff();
      if (j0 + 16 + lo > qrow) a1 = -__builtin_inff();
      float mx = fmaxf(a0, a1);
#pragma unroll
      for (int d = 1; d < 16; d <<= 1) mx = fmaxf(mx, __shfl_xor(mx, d, 32));
      const float mnew  = fmaxf(mrow[r], mx);
      const float alpha = __expf(mrow[r] - mnew);
      const float e0 = __expf(a0 - mnew);
      const float e1 = __expf(a1 - mnew);
      float sum = e0 + e1;
#pragma unroll
      for (int d = 1; d < 16; d <<= 1) sum += __shfl_xor(sum, d, 32);
      lrow[r] = lrow[r] * alpha + sum;
      mrow[r] = mnew;
      o[0][r] *= alpha; o[1][r] *= alpha; o[2][r] *= alpha; o[3][r] *= alpha;
      unsigned short* pr = &Ps[wave][(r + 8 * hi) * LSTR];
      pr[lo]      = f32_to_bf16(e0);
      pr[16 + lo] = f32_to_bf16(e1);
    }
    asm volatile("s_wait_dscnt 0" ::: "memory");  // P writes visible to re-read

    // reload P as a 16x32 bf16 A fragment
    FragB16 pa;
#pragma unroll
    for (int j = 0; j < 8; ++j) {
      const int kk = hi * 16 + j * 2;
      pa.u[j] = *(const unsigned*)(&Ps[wave][lo * LSTR + kk]);
    }
    // O += P V : 4 WMMAs over dk columns
#pragma unroll
    for (int c = 0; c < 4; ++c) {
      FragB16 vb;
      const unsigned short* base = &Vt[(c * 16 + lo) * LSTR + hi * 16];
#pragma unroll
      for (int j = 0; j < 8; ++j) vb.u[j] = *(const unsigned*)(base + 2 * j);
      o[c] = __builtin_amdgcn_wmma_f32_16x16x32_bf16(
          false, pa.v, false, vb.v, (short)0, o[c], false, false);
    }
  }

  // epilogue: normalize and store bf16 attention output
#pragma unroll
  for (int c = 0; c < 4; ++c)
#pragma unroll
    for (int r = 0; r < 8; ++r) {
      const int row = q0w + r + 8 * hi;
      const float val = o[c][r] / lrow[r];
      Obuf[(size_t)(b * S_LEN + row) * D_MODEL + h * DK + c * 16 + lo] =
          f32_to_bf16(val);
    }
}

// ---------------------------------------------------------------------------
// Host-side launch
// ---------------------------------------------------------------------------
extern "C" void kernel_launch(void* const* d_in, const int* in_sizes, int n_in,
                              void* d_out, int out_size, void* d_ws,
                              size_t ws_size, hipStream_t stream) {
  (void)in_sizes; (void)n_in; (void)out_size; (void)ws_size;
  const float* x  = (const float*)d_in[0];
  const float* Wq = (const float*)d_in[1];
  const float* Wk = (const float*)d_in[2];
  const float* Wv = (const float*)d_in[3];
  const float* Wo = (const float*)d_in[4];

  // workspace layout (bf16 buffers, 256B-aligned offsets)
  char* ws = (char*)d_ws;
  unsigned short* xb    = (unsigned short*)(ws);                       //  8 MB
  unsigned short* Wqkvb = (unsigned short*)(ws + (size_t)8  * 1024 * 1024);
  unsigned short* Wob   = (unsigned short*)(ws + (size_t)14 * 1024 * 1024);
  unsigned short* QKV   = (unsigned short*)(ws + (size_t)16 * 1024 * 1024);
  unsigned short* Obuf  = (unsigned short*)(ws + (size_t)42 * 1024 * 1024);

  const int nw = D_MODEL * D_MODEL;          // 1M elems per weight
  const int nx = NTOK * D_MODEL;             // 4M elems

  cvt_bf16_kernel<<<(nx + 255) / 256, 256, 0, stream>>>(x, xb, nx);
  cvt_bf16_kernel<<<(nw + 255) / 256, 256, 0, stream>>>(Wq, Wqkvb, nw);
  cvt_bf16_kernel<<<(nw + 255) / 256, 256, 0, stream>>>(Wk, Wqkvb + (size_t)nw, nw);
  cvt_bf16_kernel<<<(nw + 255) / 256, 256, 0, stream>>>(Wv, Wqkvb + (size_t)2 * nw, nw);
  cvt_bf16_kernel<<<(nw + 255) / 256, 256, 0, stream>>>(Wo, Wob, nw);

  // QKV[4096,3072] = xb[4096,1024] @ Wqkv[3072,1024]^T  (bf16 out)
  gemm_bf16_wT_kernel<false>
      <<<dim3(3 * D_MODEL / 128, NTOK / 128), 256, 0, stream>>>(
          xb, Wqkvb, (void*)QKV, NTOK, 3 * D_MODEL, D_MODEL);

  // causal flash attention -> Obuf[4096,1024] bf16
  flash_attn_kernel<<<dim3(S_LEN / 64, BATCH * NHEADS), 128, 0, stream>>>(
      QKV, Obuf);

  // out[4096,1024] f32 = Obuf @ Wo^T
  gemm_bf16_wT_kernel<true>
      <<<dim3(D_MODEL / 128, NTOK / 128), 256, 0, stream>>>(
          Obuf, Wob, d_out, NTOK, D_MODEL, D_MODEL);
}
